// SimpleInstanceAttention_42640435315261
// MI455X (gfx1250) — matrix operand order repaired and checked by reading, not compiled, not run on hardware
//
#include <hip/hip_runtime.h>
#include <hip/hip_bf16.h>

// ---------------------------------------------------------------------------
// Problem constants (from reference setup_inputs)
// ---------------------------------------------------------------------------
#define BATCH     8
#define L1        300
#define L2        4096        // Hf*Wf = 64*64
#define QDIM      256
#define VDIM      256
#define NHEAD     8
#define HDIM      32
#define NBOX      4
#define PP        16          // K*K sample points per box

typedef __attribute__((ext_vector_type(16))) __bf16       v16bf;
typedef __attribute__((ext_vector_type(8)))  float        v8f;
typedef __attribute__((ext_vector_type(8)))  unsigned int v8u;
typedef __attribute__((ext_vector_type(4)))  unsigned int v4u;

// fp32 x2 -> packed bf16x2 (round-to-nearest-even)
__device__ __forceinline__ unsigned int f2bf2(float lo, float hi) {
  unsigned int ul = __builtin_bit_cast(unsigned int, lo);
  unsigned int uh = __builtin_bit_cast(unsigned int, hi);
  ul += 0x7FFFu + ((ul >> 16) & 1u);
  uh += 0x7FFFu + ((uh >> 16) & 1u);
  return (ul >> 16) | (uh & 0xFFFF0000u);
}

__device__ __forceinline__ unsigned short f2bf1(float x) {
  unsigned int u = __builtin_bit_cast(unsigned int, x);
  u += 0x7FFFu + ((u >> 16) & 1u);
  return (unsigned short)(u >> 16);
}

// ---------------------------------------------------------------------------
// One-time fp32 -> bf16 conversion (vectorized: 8 floats -> 4 dwords / thread)
// ---------------------------------------------------------------------------
__global__ void cvt_bf16_kernel(const float* __restrict__ src,
                                unsigned int* __restrict__ dst, int n8) {
  const int i = blockIdx.x * blockDim.x + threadIdx.x;
  if (i >= n8) return;
  const float* p = src + (size_t)i * 8;
  v4u o;
  o[0] = f2bf2(p[0], p[1]);
  o[1] = f2bf2(p[2], p[3]);
  o[2] = f2bf2(p[4], p[5]);
  o[3] = f2bf2(p[6], p[7]);
  *(v4u*)(dst + (size_t)i * 4) = o;
}

// ---------------------------------------------------------------------------
// bf16 fragment loaders: pure b128 vector loads, no conversion VALU.
// A: 16x32, ISA layout: lanes 0-15 (row=lane) K={0..7,16..23};
//                       lanes 16-31 (row=lane-16) K={8..15,24..31}.
// ---------------------------------------------------------------------------
__device__ __forceinline__ v16bf load_a_frag(const unsigned short* __restrict__ tile,
                                             int ld) {
  const int lane = threadIdx.x & 31;
  const unsigned short* p = tile + (lane & 15) * ld + ((lane & 16) ? 8 : 0);
  v4u lo = *(const v4u*)(p);        // K {0..7}   (16B aligned)
  v4u hi = *(const v4u*)(p + 16);   // K {16..23}
  v8u u = {lo[0], lo[1], lo[2], lo[3], hi[0], hi[1], hi[2], hi[3]};
  return __builtin_bit_cast(v16bf, u);
}

// B: 32x16 with B[k][n] = W[n][k], W row-major bf16 (ld).
// lanes 0-15 (col=lane) hold K={0..7,16..23}; lanes 16-31 hold K={8..15,24..31}.
__device__ __forceinline__ v16bf load_b_frag(const unsigned short* __restrict__ tile,
                                             int ld) {
  const int lane = threadIdx.x & 31;
  const unsigned short* p = tile + (lane & 15) * ld + ((lane & 16) ? 8 : 0);
  v4u lo = *(const v4u*)(p);
  v4u hi = *(const v4u*)(p + 16);
  v8u u = {lo[0], lo[1], lo[2], lo[3], hi[0], hi[1], hi[2], hi[3]};
  return __builtin_bit_cast(v16bf, u);
}

// ---------------------------------------------------------------------------
// Kernel 1: value projection  vm[b,h,pos,c] = mask ? 0 : (value @ Wv^T + bv)
// One wave per 16x16 tile; K=256 in 8 WMMA steps, 4 b128 loads per step.
// ---------------------------------------------------------------------------
__global__ void vproj_kernel(const unsigned short* __restrict__ value_bf,
                             const unsigned short* __restrict__ Wbf,
                             const float* __restrict__ bias,
                             const unsigned char* __restrict__ mask,
                             float* __restrict__ vm) {
  const int wave = blockIdx.x * (blockDim.x >> 5) + (threadIdx.x >> 5);
  const int nt = wave & 15;                 // 16 column tiles
  const int mt = wave >> 4;                 // 2048 row tiles
  if (mt >= (BATCH * L2) / 16) return;
  const int lane = threadIdx.x & 31;

  const unsigned short* Abase = value_bf + (size_t)(mt * 16) * VDIM;
  const unsigned short* Bbase = Wbf + (size_t)(nt * 16) * VDIM;
  v8f acc = {};
#pragma unroll
  for (int kk = 0; kk < VDIM; kk += 32) {
    __builtin_prefetch(Abase + kk + 128, 0, 1);          // global_prefetch_b8
    v16bf a = load_a_frag(Abase + kk, VDIM);
    v16bf b = load_b_frag(Bbase + kk, VDIM);
    acc = __builtin_amdgcn_wmma_f32_16x16x32_bf16(false, a, false, b,
                                                  (short)0, acc, false, false);
  }
  const int col = nt * 16 + (lane & 15);
  const int h = col >> 5, c = col & 31;
  const int rbase = mt * 16 + ((lane & 16) ? 8 : 0);
  const float bn = bias[col];
#pragma unroll
  for (int i = 0; i < 8; ++i) {
    const int m = rbase + i;
    const int bb = m >> 12, pos = m & (L2 - 1);
    float v = acc[i] + bn;
    if (mask[m]) v = 0.0f;
    vm[(((size_t)(bb * NHEAD + h) * L2 + pos) * HDIM) + c] = v;
  }
}

// ---------------------------------------------------------------------------
// Kernel 2a: query projections (box offsets + attention logits), fused.
// ---------------------------------------------------------------------------
__global__ void qproj_kernel(const unsigned short* __restrict__ query_bf,
                             const unsigned short* __restrict__ wbox_bf,
                             const float* __restrict__ box_b,
                             const unsigned short* __restrict__ wattn_bf,
                             const float* __restrict__ attn_b,
                             float* __restrict__ off_raw,
                             float* __restrict__ aw_raw) {
  const int wave = blockIdx.x * (blockDim.x >> 5) + (threadIdx.x >> 5);
  if (wave >= (BATCH * L1 / 16) * 16) return;
  const int nt = wave & 15;
  const int mt = wave >> 4;
  const bool is_attn = nt >= 8;
  const int ntl = nt & 7;
  const unsigned short* W = is_attn ? wattn_bf : wbox_bf;
  const float* Bv = is_attn ? attn_b : box_b;
  float* dst = is_attn ? aw_raw : off_raw;
  const int lane = threadIdx.x & 31;

  const unsigned short* Abase = query_bf + (size_t)(mt * 16) * QDIM;
  const unsigned short* Bbase = W + (size_t)(ntl * 16) * QDIM;
  v8f acc = {};
#pragma unroll
  for (int kk = 0; kk < QDIM; kk += 32) {
    v16bf a = load_a_frag(Abase + kk, QDIM);
    v16bf b = load_b_frag(Bbase + kk, QDIM);
    acc = __builtin_amdgcn_wmma_f32_16x16x32_bf16(false, a, false, b,
                                                  (short)0, acc, false, false);
  }
  const int col = ntl * 16 + (lane & 15);
  const int rbase = mt * 16 + ((lane & 16) ? 8 : 0);
  const float bn = Bv[col];
#pragma unroll
  for (int i = 0; i < 8; ++i)
    dst[(size_t)(rbase + i) * 128 + col] = acc[i] + bn;
}

// ---------------------------------------------------------------------------
// Kernel 2b: per-(b,l,h): sampling grid, spatial/level softmaxes, aw output.
// ---------------------------------------------------------------------------
__global__ void prep_kernel(const float* __restrict__ off_raw,
                            const float* __restrict__ aw_raw,
                            const float* __restrict__ ref_windows,
                            const float* __restrict__ vr,
                            float* __restrict__ gridbuf,
                            float* __restrict__ spatial,
                            float* __restrict__ level,
                            float* __restrict__ aw_out) {
  const int t = blockIdx.x * blockDim.x + threadIdx.x;
  if (t >= BATCH * L1 * NHEAD) return;
  const int h = t & 7;
  const int bl = t >> 3;                    // b*L1 + l
  const int b = bl / L1;

  const float* refp = ref_windows + bl * 4;
  const float cx = refp[0], cy = refp[1], rw = refp[2], rh = refp[3];
  const float rx = vr[b * 2 + 0], ry = vr[b * 2 + 1];
  const float* offp = off_raw + (size_t)bl * 128 + h * 16;
  const float* awp  = aw_raw  + (size_t)bl * 128 + h * 16;

  float raw[16], e[16];
  float mx = -1e30f;
#pragma unroll
  for (int i = 0; i < 16; ++i) { raw[i] = awp[i]; mx = fmaxf(mx, raw[i]); }
  float S = 0.f;
#pragma unroll
  for (int i = 0; i < 16; ++i) { e[i] = __expf(raw[i] - mx); S += e[i]; }
  float Lq[4] = {0.f, 0.f, 0.f, 0.f};
#pragma unroll
  for (int box = 0; box < NBOX; ++box)
#pragma unroll
    for (int q = 0; q < 4; ++q) Lq[q] += e[box * 4 + q];
  const float invS = 1.0f / (4.0f * S);     // each raw value appears 4x expanded
  float invL[4];
#pragma unroll
  for (int q = 0; q < 4; ++q) invL[q] = 1.0f / Lq[q];

  float* gp = gridbuf + (size_t)t * (NBOX * PP * 2);
  float* sp = spatial + (size_t)t * (NBOX * PP);
  float* lp = level   + (size_t)t * (NBOX * PP);
  float* ap = aw_out  + (size_t)t * (NBOX * PP);

#pragma unroll
  for (int box = 0; box < NBOX; ++box) {
    const float o0 = offp[box * 4 + 0], o1 = offp[box * 4 + 1];
    const float o2 = offp[box * 4 + 2], o3 = offp[box * 4 + 3];
    const float bx = cx + o0 * 0.125f * rw;
    const float by = cy + o1 * 0.125f * rh;
    const float bw = rw + o2 * 0.125f * rw;
    const float bh = rh + o3 * 0.125f * rh;
    const float sxs = fmaxf(bw, 0.f), sys = fmaxf(bh, 0.f);   // relu(size)
#pragma unroll
    for (int I = 0; I < 4; ++I)
#pragma unroll
      for (int J = 0; J < 4; ++J) {
        const int p = I * 4 + J;
        const float kx = ((float)J - 1.5f) * 0.25f;
        const float ky = ((float)I - 1.5f) * 0.25f;
        gp[(box * PP + p) * 2 + 0] = (bx + kx * sxs) * rx;
        gp[(box * PP + p) * 2 + 1] = (by + ky * sys) * ry;
        const int q = (I >> 1) * 2 + (J >> 1);
        const float ev = e[box * 4 + q];
        sp[box * PP + p] = ev * invS;
        lp[box * PP + p] = ev * invL[q];
        ap[box * PP + p] = raw[box * 4 + q];   // expanded aw (3rd output)
      }
  }
}

// ---------------------------------------------------------------------------
// Kernel 3: bilinear sampling + weighted reduction over the 64 points.
// One wave per (b,l,h); lane = channel. vm (33.5 MB) lives in the 192 MB L2.
// Results stored as bf16 so the output-projection GEMM reads them directly.
// ---------------------------------------------------------------------------
__global__ void sample_kernel(const float* __restrict__ vm,
                              const float* __restrict__ gridbuf,
                              const float* __restrict__ spatial,
                              const float* __restrict__ level,
                              const long long* __restrict__ vshape,
                              unsigned short* __restrict__ outbuf_bf,
                              unsigned short* __restrict__ moutbuf_bf) {
  const int wave = blockIdx.x * (blockDim.x >> 5) + (threadIdx.x >> 5);
  if (wave >= BATCH * L1 * NHEAD) return;
  const int lane = threadIdx.x & 31;
  const int h = wave & 7;
  const int bl = wave >> 3;
  const int b = bl / L1;
  const int Hf = (int)vshape[0];
  const int Wf = (int)vshape[1];
  const float Hff = (float)Hf, Wff = (float)Wf;

  const float* vbase = vm + (size_t)(b * NHEAD + h) * (size_t)(Hf * Wf) * HDIM;
  const float* gp = gridbuf + (size_t)wave * (NBOX * PP * 2);
  const float* sp = spatial + (size_t)wave * (NBOX * PP);
  const float* lp = level   + (size_t)wave * (NBOX * PP);

  float ao = 0.f, am = 0.f;
  for (int p = 0; p < NBOX * PP; ++p) {
    const float x = gp[p * 2 + 0] * Wff - 0.5f;
    const float y = gp[p * 2 + 1] * Hff - 0.5f;
    const float x0f = floorf(x), y0f = floorf(y);
    const float wx1 = x - x0f, wy1 = y - y0f;
    const float wx0 = 1.f - wx1, wy0 = 1.f - wy1;
    const int x0 = (int)x0f, y0 = (int)y0f;
    float s = 0.f;
#pragma unroll
    for (int cyi = 0; cyi < 2; ++cyi) {
#pragma unroll
      for (int cxi = 0; cxi < 2; ++cxi) {
        const int xi = x0 + cxi, yi = y0 + cyi;
        const float w = (cxi ? wx1 : wx0) * (cyi ? wy1 : wy0);
        if (xi >= 0 && xi < Wf && yi >= 0 && yi < Hf)
          s += w * vbase[(size_t)(yi * Wf + xi) * HDIM + lane];
      }
    }
    ao += sp[p] * s;
    am += lp[p] * s;
  }
  const size_t oidx = (size_t)bl * VDIM + h * HDIM + lane;
  outbuf_bf [oidx] = f2bf1(ao);
  moutbuf_bf[oidx] = f2bf1(am);
}

// ---------------------------------------------------------------------------
// Kernel 4: output projections (output and mask_output), WMMA bf16.
// ---------------------------------------------------------------------------
__global__ void oproj_kernel(const unsigned short* __restrict__ outbuf_bf,
                             const unsigned short* __restrict__ moutbuf_bf,
                             const unsigned short* __restrict__ Wbf,
                             const float* __restrict__ bias,
                             float* __restrict__ out) {
  const int wave = blockIdx.x * (blockDim.x >> 5) + (threadIdx.x >> 5);
  const int ntiles = (BATCH * L1 / 16) * 16;      // 2400 tiles per matrix
  if (wave >= 2 * ntiles) return;
  const int sel = wave >= ntiles;
  const int w2 = wave - sel * ntiles;
  const int nt = w2 & 15;
  const int mt = w2 >> 4;
  const unsigned short* src = sel ? moutbuf_bf : outbuf_bf;
  float* dst = out + (size_t)sel * (BATCH * L1 * QDIM);
  const int lane = threadIdx.x & 31;

  const unsigned short* Abase = src + (size_t)(mt * 16) * VDIM;
  const unsigned short* Bbase = Wbf + (size_t)(nt * 16) * VDIM;
  v8f acc = {};
#pragma unroll
  for (int kk = 0; kk < VDIM; kk += 32) {
    v16bf a = load_a_frag(Abase + kk, VDIM);
    v16bf b = load_b_frag(Bbase + kk, VDIM);
    acc = __builtin_amdgcn_wmma_f32_16x16x32_bf16(false, a, false, b,
                                                  (short)0, acc, false, false);
  }
  const int col = nt * 16 + (lane & 15);
  const int rbase = mt * 16 + ((lane & 16) ? 8 : 0);
  const float bn = bias[col];
#pragma unroll
  for (int i = 0; i < 8; ++i)
    dst[(size_t)(rbase + i) * QDIM + col] = acc[i] + bn;
}

// ---------------------------------------------------------------------------
// Launch
// ---------------------------------------------------------------------------
extern "C" void kernel_launch(void* const* d_in, const int* in_sizes, int n_in,
                              void* d_out, int out_size, void* d_ws, size_t ws_size,
                              hipStream_t stream) {
  const float*         query   = (const float*)d_in[0];
  const float*         value   = (const float*)d_in[1];
  const long long*     v_shape = (const long long*)d_in[2];
  const unsigned char* v_mask  = (const unsigned char*)d_in[3];
  /* d_in[4] = v_start_index (unused) */
  const float* vr          = (const float*)d_in[5];
  const float* ref_windows = (const float*)d_in[6];
  const float* box_w   = (const float*)d_in[7];
  const float* box_b   = (const float*)d_in[8];
  const float* attn_w  = (const float*)d_in[9];
  const float* attn_b  = (const float*)d_in[10];
  const float* vproj_w = (const float*)d_in[11];
  const float* vproj_b = (const float*)d_in[12];
  const float* oproj_w = (const float*)d_in[13];
  const float* oproj_b = (const float*)d_in[14];

  float* out = (float*)d_out;

  // ---- workspace carve (256B aligned regions) ----
  char* wsp = (char*)d_ws;
  auto carve = [&](size_t bytes) -> char* {
    char* p = wsp;
    wsp += (bytes + 255) & ~(size_t)255;
    return p;
  };
  float*          vm         = (float*)         carve((size_t)BATCH * NHEAD * L2 * HDIM * 4);   // 33.5 MB
  unsigned short* value_bf   = (unsigned short*)carve((size_t)BATCH * L2 * VDIM * 2);           // 16.8 MB
  unsigned short* query_bf   = (unsigned short*)carve((size_t)BATCH * L1 * QDIM * 2);
  unsigned short* wv_bf      = (unsigned short*)carve((size_t)VDIM * VDIM * 2);
  unsigned short* wo_bf      = (unsigned short*)carve((size_t)QDIM * VDIM * 2);
  unsigned short* wbox_bf    = (unsigned short*)carve((size_t)128 * QDIM * 2);
  unsigned short* wattn_bf   = (unsigned short*)carve((size_t)128 * QDIM * 2);
  float*          off_raw    = (float*)         carve((size_t)BATCH * L1 * 128 * 4);
  float*          aw_raw     = (float*)         carve((size_t)BATCH * L1 * 128 * 4);
  float*          gridbuf    = (float*)         carve((size_t)BATCH * L1 * NHEAD * NBOX * PP * 2 * 4);
  float*          spatialw   = (float*)         carve((size_t)BATCH * L1 * NHEAD * NBOX * PP * 4);
  float*          levelw     = (float*)         carve((size_t)BATCH * L1 * NHEAD * NBOX * PP * 4);
  unsigned short* outbuf_bf  = (unsigned short*)carve((size_t)BATCH * L1 * VDIM * 2);
  unsigned short* moutbuf_bf = (unsigned short*)carve((size_t)BATCH * L1 * VDIM * 2);

  // d_out: output[614400] | mask_output[614400] | aw[1228800]
  float* out_main = out;
  float* out_aw   = out + 2 * (BATCH * L1 * QDIM);

  // 0) one-time fp32 -> bf16 conversions (memory bound, ~8 us total)
  {
    int n8;
    n8 = (BATCH * L2 * VDIM) / 8;      // 1,048,576
    cvt_bf16_kernel<<<(n8 + 255) / 256, 256, 0, stream>>>(value, (unsigned int*)value_bf, n8);
    n8 = (BATCH * L1 * QDIM) / 8;      // 76,800
    cvt_bf16_kernel<<<(n8 + 255) / 256, 256, 0, stream>>>(query, (unsigned int*)query_bf, n8);
    n8 = (VDIM * VDIM) / 8;
    cvt_bf16_kernel<<<(n8 + 255) / 256, 256, 0, stream>>>(vproj_w, (unsigned int*)wv_bf, n8);
    cvt_bf16_kernel<<<(n8 + 255) / 256, 256, 0, stream>>>(oproj_w, (unsigned int*)wo_bf, n8);
    n8 = (128 * QDIM) / 8;
    cvt_bf16_kernel<<<(n8 + 255) / 256, 256, 0, stream>>>(box_w, (unsigned int*)wbox_bf, n8);
    cvt_bf16_kernel<<<(n8 + 255) / 256, 256, 0, stream>>>(attn_w, (unsigned int*)wattn_bf, n8);
  }

  // 1) value projection: 32768 tiles, 4 waves/block
  vproj_kernel<<<8192, 128, 0, stream>>>(value_bf, wv_bf, vproj_b, v_mask, vm);

  // 2a) query projections: 2400 tiles
  qproj_kernel<<<600, 128, 0, stream>>>(query_bf, wbox_bf, box_b, wattn_bf, attn_b,
                                        off_raw, aw_raw);

  // 2b) grid / softmax prep: 19200 threads
  prep_kernel<<<75, 256, 0, stream>>>(off_raw, aw_raw, ref_windows, vr,
                                      gridbuf, spatialw, levelw, out_aw);

  // 3) bilinear sampling + reduction: 19200 waves, 8 waves/block
  sample_kernel<<<2400, 256, 0, stream>>>(vm, gridbuf, spatialw, levelw,
                                          v_shape, outbuf_bf, moutbuf_bf);

  // 4) output projections: 4800 tiles
  oproj_kernel<<<1200, 128, 0, stream>>>(outbuf_bf, moutbuf_bf, wo_bf, oproj_b,
                                         out_main);
}
